// DPAB_5927054868613
// MI455X (gfx1250) — matrix operand order repaired and checked by reading, not compiled, not run on hardware
//
#include <hip/hip_runtime.h>
#include <hip/hip_bf16.h>

typedef __attribute__((ext_vector_type(16))) __bf16 v16bf;
typedef __attribute__((ext_vector_type(8)))  __bf16 v8bf;
typedef __attribute__((ext_vector_type(8)))  float  v8f;
typedef __attribute__((ext_vector_type(4)))  float  f32x4;

#define BATCH   8
#define HH      96
#define WW      96
#define LTOT    (HH*WW)        /* 9216 */
#define DMODEL  192
#define DINNER  384
#define DSTATE  64
#define CONVD   448            /* DINNER + DSTATE */
#define DPROJ   832            /* 2*DINNER + DSTATE */
#define NHEADS  6
#define HEADDIM 64
#define SCALEF  0.125f         /* 64^-0.5 */

#define LDSTRIDE 40            /* padded bf16 row stride: 80B, 16B-aligned, */
                               /* 20-bank stride -> conflict-free frag reads */

// ---------------------------------------------------------------------------
// Batched GEMM: C = alpha * op(A) * op(B) (+ bias[n]).
// fp32 in/out, bf16 WMMA (v_wmma_f32_16x16x32_bf16), f32 accumulation.
// Block = 4 waves sharing one 64-wide N group; each wave owns a 16x64 C tile
// (4 WMMAs per K-step from one A fragment). The 32Kx64N B tile (and the A
// tile when AT=1) is staged in LDS in bf16 fragment order, DOUBLE-BUFFERED:
// one __syncthreads per K-step; global fetch of step k+1 issues before the
// WMMAs of step k (raw floats parked in registers until after the WMMAs), so
// HBM latency hides behind matrix math. All 4 B fragments are loaded from
// LDS before any WMMA so ds waits amortize.
// AT: A[m,k] = Amem[k*lda+m]; BT: B[k,n] = Bmem[n*ldb+k].
// Batch offset per operand: off = (by/hdiv)*Out + (by%hdiv)*In.
// Requirements: M%16==0, (M/16)%4==0, N%64==0, K%32==0; grid.x = tiles/4
// exactly (no inactive waves -> EXEC all-ones at WMMA, barriers safe).
// ---------------------------------------------------------------------------
template<int AT, int BT, int HASBIAS>
__global__ __launch_bounds__(128)
void gemm_wmma(const float* __restrict__ A, const float* __restrict__ B,
               float* __restrict__ C, const float* __restrict__ bias,
               int M, int N, int K, int lda, int ldb, int ldc, int hdiv,
               long aOut, long aIn, long bOut, long bIn, long cOut, long cIn,
               float alpha)
{
    __shared__ __bf16 lds_b[2][64 * LDSTRIDE];
    __shared__ __bf16 lds_a[(AT == 1) ? 2 : 1][(AT == 1) ? 64 * LDSTRIDE : 8];

    int by = blockIdx.y;
    int b0 = by / hdiv, b1 = by % hdiv;
    A += b0 * aOut + b1 * aIn;
    B += b0 * bOut + b1 * bIn;
    C += b0 * cOut + b1 * cIn;

    int tilesM = M >> 4;
    int wave = threadIdx.x >> 5;
    int lane = threadIdx.x & 31;
    int tile = blockIdx.x * 4 + wave;
    int m0 = (tile % tilesM) << 4;
    int n0 = (tile / tilesM) << 6;           // shared by all 4 waves
    int blockM0 = ((blockIdx.x * 4) % tilesM) << 4;  // staged A row base (AT=1)

    int half = lane >> 4;                    // 0: lanes 0-15, 1: lanes 16-31
    int l15  = lane & 15;
    int rowA = m0 + l15;

    // cooperative-fill coordinates: thread -> (row 0..63, k-half 0..1)
    int cN  = threadIdx.x >> 1;
    int cK0 = (threadIdx.x & 1) * 16;

    // ---- register-staged tile fetch / LDS store helpers ----
    auto fetchB = [&](int kb, f32x4* r) {
        if (BT == 1) {
            const f32x4* src = (const f32x4*)(B + (long)(n0 + cN) * ldb + kb + cK0);
            r[0] = src[0]; r[1] = src[1]; r[2] = src[2]; r[3] = src[3];
        } else {
            const float* src = B + (long)(kb + cK0) * ldb + n0 + cN;
            #pragma unroll
            for (int j = 0; j < 4; ++j)
                #pragma unroll
                for (int i = 0; i < 4; ++i)
                    r[j][i] = src[(long)(4 * j + i) * ldb];
        }
    };
    auto storeB = [&](int buf, const f32x4* r) {
        v8bf w0, w1;
        #pragma unroll
        for (int j = 0; j < 4; ++j) {
            w0[j]     = (__bf16)r[0][j];
            w0[4 + j] = (__bf16)r[1][j];
            w1[j]     = (__bf16)r[2][j];
            w1[4 + j] = (__bf16)r[3][j];
        }
        *(v8bf*)&lds_b[buf][cN * LDSTRIDE + cK0]     = w0;
        *(v8bf*)&lds_b[buf][cN * LDSTRIDE + cK0 + 8] = w1;
    };
    auto fetchAT = [&](int kb, f32x4* r) {          // AT==1: A k-major tile
        const float* src = A + (long)(kb + cK0) * lda + blockM0 + cN;
        #pragma unroll
        for (int j = 0; j < 4; ++j)
            #pragma unroll
            for (int i = 0; i < 4; ++i)
                r[j][i] = src[(long)(4 * j + i) * lda];
    };
    auto storeAT = [&](int buf, const f32x4* r) {
        v8bf w0, w1;
        #pragma unroll
        for (int j = 0; j < 4; ++j) {
            w0[j]     = (__bf16)r[0][j];
            w0[4 + j] = (__bf16)r[1][j];
            w1[j]     = (__bf16)r[2][j];
            w1[4 + j] = (__bf16)r[3][j];
        }
        *(v8bf*)&lds_a[buf][cN * LDSTRIDE + cK0]     = w0;
        *(v8bf*)&lds_a[buf][cN * LDSTRIDE + cK0 + 8] = w1;
    };
    auto fetchA0 = [&](int kb, f32x4* r) {          // AT==0: per-wave rows
        const f32x4* ap = (const f32x4*)(A + (long)rowA * lda + kb + half * 8);
        r[0] = ap[0]; r[1] = ap[1]; r[2] = ap[4]; r[3] = ap[5];
    };

    v8f acc[4] = {};
    f32x4 ra[4], rb[4];

    // prologue: stage K-step 0
    fetchB(0, rb);
    if (AT == 1) fetchAT(0, ra); else fetchA0(0, ra);
    storeB(0, rb);
    if (AT == 1) storeAT(0, ra);

    for (int kb = 0; kb < K; kb += 32) {
        int cur = (kb >> 5) & 1;
        __syncthreads();                     // publish buf[cur]; fence reads
        bool more = (kb + 32) < K;
        f32x4 na[4], nb[4];
        if (more) {                          // issue next tile's global loads
            fetchB(kb + 32, nb);
            if (AT == 1) fetchAT(kb + 32, na); else fetchA0(kb + 32, na);
        }

        // ---- A fragment: lanes0-15 K=kb+{0..7,16..23},
        //                  lanes16-31 K=kb+{8..15,24..31}
        v16bf af;
        if (AT == 0) {
            #pragma unroll
            for (int j = 0; j < 4; ++j) {
                af[j]      = (__bf16)ra[0][j];
                af[4 + j]  = (__bf16)ra[1][j];
                af[8 + j]  = (__bf16)ra[2][j];
                af[12 + j] = (__bf16)ra[3][j];
            }
        } else {
            int rloc = (wave << 4) + l15;
            v8bf alo = *(const v8bf*)&lds_a[cur][rloc * LDSTRIDE + half * 8];
            v8bf ahi = *(const v8bf*)&lds_a[cur][rloc * LDSTRIDE + half * 8 + 16];
            af = __builtin_shufflevector(alo, ahi,
                                         0,1,2,3,4,5,6,7,8,9,10,11,12,13,14,15);
        }
        // ---- all 4 B fragments first (ds waits amortize), then 4 WMMAs ----
        v16bf bfr[4];
        #pragma unroll
        for (int f = 0; f < 4; ++f) {
            int nloc = l15 + f * 16;
            v8bf blo = *(const v8bf*)&lds_b[cur][nloc * LDSTRIDE + half * 8];
            v8bf bhi = *(const v8bf*)&lds_b[cur][nloc * LDSTRIDE + half * 8 + 16];
            bfr[f] = __builtin_shufflevector(blo, bhi,
                                         0,1,2,3,4,5,6,7,8,9,10,11,12,13,14,15);
        }
        #pragma unroll
        for (int f = 0; f < 4; ++f)
            acc[f] = __builtin_amdgcn_wmma_f32_16x16x32_bf16(
                false, af, false, bfr[f], (short)0, acc[f], false, false);

        if (more) {                          // park next tile into buf[cur^1]
            storeB(cur ^ 1, nb);
            if (AT == 1) storeAT(cur ^ 1, na);
            #pragma unroll
            for (int j = 0; j < 4; ++j) ra[j] = na[j];
        }
    }

    // C/D f32 16x16: VGPR r -> M=m0+r (lanes0-15) / M=m0+r+8 (lanes16-31)
    #pragma unroll
    for (int f = 0; f < 4; ++f) {
        int colf = n0 + l15 + f * 16;
        float bb = HASBIAS ? bias[colf] : 0.f;
        #pragma unroll
        for (int r = 0; r < 8; ++r) {
            int rr = m0 + r + half * 8;
            C[(long)rr * ldc + colf] = acc[f][r] * alpha + bb;
        }
    }
}

template<int AT, int BT, int HB>
static void run_gemm(hipStream_t s,
                     const float* A, const float* B, float* C, const float* bias,
                     int M, int N, int K, int lda, int ldb, int ldc,
                     int batches, int hdiv,
                     long aOut, long aIn, long bOut, long bIn, long cOut, long cIn,
                     float alpha)
{
    int tiles = (M >> 4) * (N >> 6);        // always a multiple of 4 here
    dim3 grid((unsigned)(tiles / 4), (unsigned)batches);
    gemm_wmma<AT, BT, HB><<<grid, 128, 0, s>>>(A, B, C, bias, M, N, K,
                                               lda, ldb, ldc, hdiv,
                                               aOut, aIn, bOut, bIn, cOut, cIn,
                                               alpha);
}

// ---------------------------------------------------------------------------
// 3x3 depthwise conv (pad 1) over the xB slice of zxB (channels 384..831),
// + bias + SiLU. Channel-last, coalesced over c. out: (b, L, 448)
// ---------------------------------------------------------------------------
__global__ void conv3_silu_kernel(const float* __restrict__ zxB,
                                  const float* __restrict__ w,
                                  const float* __restrict__ bias,
                                  float* __restrict__ out)
{
    long idx = (long)blockIdx.x * blockDim.x + threadIdx.x;
    const long total = (long)BATCH * LTOT * CONVD;
    if (idx >= total) return;
    int c = (int)(idx % CONVD);
    long bl = idx / CONVD;
    int l = (int)(bl % LTOT);
    int b = (int)(bl / LTOT);
    int h = l / WW, x = l % WW;
    float acc = bias[c];
    #pragma unroll
    for (int dy = -1; dy <= 1; ++dy) {
        int hh = h + dy;
        if (hh < 0 || hh >= HH) continue;
        #pragma unroll
        for (int dx = -1; dx <= 1; ++dx) {
            int ww = x + dx;
            if (ww < 0 || ww >= WW) continue;
            acc += w[c * 9 + (dy + 1) * 3 + (dx + 1)] *
                   zxB[((long)b * LTOT + (long)hh * WW + ww) * DPROJ + DINNER + c];
        }
    }
    out[idx] = acc / (1.f + __expf(-acc));   // SiLU
}

// ---------------------------------------------------------------------------
// 5x5 depthwise conv (pad 2) over x (first 384 channels of XBC) + bias.
// out: (b, L, 384)
// ---------------------------------------------------------------------------
__global__ void conv5_kernel(const float* __restrict__ xbc,
                             const float* __restrict__ w,
                             const float* __restrict__ bias,
                             float* __restrict__ out)
{
    long idx = (long)blockIdx.x * blockDim.x + threadIdx.x;
    const long total = (long)BATCH * LTOT * DINNER;
    if (idx >= total) return;
    int c = (int)(idx % DINNER);
    long bl = idx / DINNER;
    int l = (int)(bl % LTOT);
    int b = (int)(bl / LTOT);
    int h = l / WW, x = l % WW;
    float acc = bias[c];
    #pragma unroll
    for (int dy = -2; dy <= 2; ++dy) {
        int hh = h + dy;
        if (hh < 0 || hh >= HH) continue;
        #pragma unroll
        for (int dx = -2; dx <= 2; ++dx) {
            int ww = x + dx;
            if (ww < 0 || ww >= WW) continue;
            acc += w[c * 25 + (dy + 2) * 5 + (dx + 2)] *
                   xbc[((long)b * LTOT + (long)hh * WW + ww) * CONVD + c];
        }
    }
    out[idx] = acc;
}

// ---------------------------------------------------------------------------
// K_soft: sigmoid(Bm) / (sum + 1e-6) over DSTATE=64. One wave32 per (b,l)
// row, 2 elements per lane, shfl_xor reduction.
// ---------------------------------------------------------------------------
__global__ void ksoft_kernel(const float* __restrict__ xbc, float* __restrict__ ks)
{
    int gid = blockIdx.x * blockDim.x + threadIdx.x;
    int row = gid >> 5;
    int lane = gid & 31;
    if (row >= BATCH * LTOT) return;
    const float* p = xbc + (long)row * CONVD + DINNER;
    float s0 = 1.f / (1.f + __expf(-p[lane]));
    float s1 = 1.f / (1.f + __expf(-p[lane + 32]));
    float sum = s0 + s1;
    #pragma unroll
    for (int m = 16; m >= 1; m >>= 1) sum += __shfl_xor(sum, m, 32);
    float inv = 1.f / (sum + 1e-6f);
    ks[(long)row * 64 + lane]      = s0 * inv;
    ks[(long)row * 64 + lane + 32] = s1 * inv;
}

// ---------------------------------------------------------------------------
// Row softmax over L=9216; one 256-thread block per (b,h,s) row, in place.
// ---------------------------------------------------------------------------
__global__ __launch_bounds__(256)
void softmax_kernel(float* __restrict__ sc)
{
    __shared__ float red[256];
    float* p = sc + (long)blockIdx.x * LTOT;
    int tid = threadIdx.x;
    float m = -3.4e38f;
    for (int i = tid; i < LTOT; i += 256) m = fmaxf(m, p[i]);
    red[tid] = m; __syncthreads();
    for (int s = 128; s >= 1; s >>= 1) {
        if (tid < s) red[tid] = fmaxf(red[tid], red[tid + s]);
        __syncthreads();
    }
    m = red[0]; __syncthreads();
    float sum = 0.f;
    for (int i = tid; i < LTOT; i += 256) {
        float e = __expf(p[i] - m); p[i] = e; sum += e;
    }
    red[tid] = sum; __syncthreads();
    for (int s = 128; s >= 1; s >>= 1) {
        if (tid < s) red[tid] += red[tid + s];
        __syncthreads();
    }
    float inv = 1.f / red[0];
    for (int i = tid; i < LTOT; i += 256) p[i] *= inv;
}

// ---------------------------------------------------------------------------
// Fused epilogue: y = fine + V*D[head]; LayerNorm(384) * g + b; * z; in place
// on T1. One 128-thread block per (b,l) row, 3 channels per thread.
// ---------------------------------------------------------------------------
__global__ __launch_bounds__(128)
void ln_mul_kernel(float* __restrict__ t1, const float* __restrict__ xbc,
                   const float* __restrict__ zxb, const float* __restrict__ Dp,
                   const float* __restrict__ g, const float* __restrict__ bln)
{
    __shared__ float red[128];
    long row = blockIdx.x;
    int tid = threadIdx.x;
    float* f = t1 + row * DINNER;
    const float* xr = xbc + row * CONVD;
    const float* zr = zxb + row * DPROJ;
    float y[3];
    float sum = 0.f;
    #pragma unroll
    for (int j = 0; j < 3; ++j) {
        int c = tid + j * 128;
        y[j] = f[c] + xr[c] * Dp[c >> 6];
        sum += y[j];
    }
    red[tid] = sum; __syncthreads();
    for (int s = 64; s >= 1; s >>= 1) {
        if (tid < s) red[tid] += red[tid + s];
        __syncthreads();
    }
    float mu = red[0] * (1.f / DINNER);
    __syncthreads();
    float sq = 0.f;
    #pragma unroll
    for (int j = 0; j < 3; ++j) { float d = y[j] - mu; sq += d * d; }
    red[tid] = sq; __syncthreads();
    for (int s = 64; s >= 1; s >>= 1) {
        if (tid < s) red[tid] += red[tid + s];
        __syncthreads();
    }
    float rs = rsqrtf(red[0] * (1.f / DINNER) + 1e-5f);
    #pragma unroll
    for (int j = 0; j < 3; ++j) {
        int c = tid + j * 128;
        float v = (y[j] - mu) * rs * g[c] + bln[c];
        f[c] = v * zr[c];
    }
}

// ---------------------------------------------------------------------------
extern "C" void kernel_launch(void* const* d_in, const int* in_sizes, int n_in,
                              void* d_out, int out_size, void* d_ws, size_t ws_size,
                              hipStream_t stream)
{
    (void)in_sizes; (void)n_in; (void)out_size; (void)ws_size;
    const float* u     = (const float*)d_in[0];
    const float* W_in  = (const float*)d_in[1];
    const float* convw = (const float*)d_in[2];
    const float* convb = (const float*)d_in[3];
    const float* dww   = (const float*)d_in[4];
    const float* dwb   = (const float*)d_in[5];
    const float* pww   = (const float*)d_in[6];
    const float* pwb   = (const float*)d_in[7];
    const float* Dp    = (const float*)d_in[8];
    const float* lng   = (const float*)d_in[9];
    const float* lnb   = (const float*)d_in[10];
    const float* W_out = (const float*)d_in[11];
    float* out = (float*)d_out;

    float* ws  = (float*)d_ws;
    float* ZXB = ws;                                        // (b,L,832)
    float* XBC = ZXB + (size_t)BATCH * LTOT * DPROJ;        // (b,L,448)
    float* KS  = XBC + (size_t)BATCH * LTOT * CONVD;        // (b,L,64)
    float* VS  = KS  + (size_t)BATCH * LTOT * DSTATE;       // (b,L,384) channel-last Vs
    float* T1  = VS  + (size_t)BATCH * LTOT * DINNER;       // (b,L,384) reused: dw5/scores/fine/y
    float* PR  = T1  + (size_t)BATCH * LTOT * DINNER;       // (b,64,384) proto
    float* RF  = PR  + (size_t)BATCH * 64 * DINNER;         // (b,64,384) refined

    const int ML = BATCH * LTOT;   // 73728

    // 1. zxB = u @ W_in   (ML x 832, K=192)
    run_gemm<0,0,0>(stream, u, W_in, ZXB, nullptr, ML, DPROJ, DMODEL,
                    DMODEL, DPROJ, DPROJ, 1, 1, 0,0, 0,0, 0,0, 1.f);

    // 2. 3x3 depthwise conv + SiLU on xB -> XBC
    {
        long total = (long)ML * CONVD;
        conv3_silu_kernel<<<(unsigned)((total + 255) / 256), 256, 0, stream>>>(
            ZXB, convw, convb, XBC);
    }

    // 3. K_soft
    ksoft_kernel<<<(ML * 32 + 255) / 256, 256, 0, stream>>>(XBC, KS);

    // 4. 5x5 depthwise conv on x -> T1
    {
        long total = (long)ML * DINNER;
        conv5_kernel<<<(unsigned)((total + 255) / 256), 256, 0, stream>>>(
            XBC, dww, dwb, T1);
    }

    // 5. pointwise: VS = T1 @ pww^T + pwb   (B stored (o,c) -> BT=1, bias)
    run_gemm<0,1,1>(stream, T1, pww, VS, pwb, ML, DINNER, DINNER,
                    DINNER, DINNER, DINNER, 1, 1, 0,0, 0,0, 0,0, 1.f);

    // 6. proto: PR[b] = KS[b]^T (64xL) @ VS[b] (Lx384)   (heads folded in N)
    run_gemm<1,0,0>(stream, KS, VS, PR, nullptr, 64, DINNER, LTOT,
                    DSTATE, DINNER, DINNER, BATCH, 1,
                    (long)LTOT * DSTATE, 0, (long)LTOT * DINNER, 0,
                    64l * DINNER, 0, 1.f);

    // 7. scores: T1[b,h] = SCALE * PR[b,:,h*64:+64] @ VS[b,:,h*64:+64]^T
    run_gemm<0,1,0>(stream, PR, VS, T1, nullptr, 64, LTOT, HEADDIM,
                    DINNER, DINNER, LTOT, BATCH * NHEADS, NHEADS,
                    64l * DINNER, (long)HEADDIM,
                    (long)LTOT * DINNER, (long)HEADDIM,
                    (long)NHEADS * 64 * LTOT, 64l * LTOT, SCALEF);

    // 8. softmax over L
    softmax_kernel<<<BATCH * NHEADS * 64, 256, 0, stream>>>(T1);

    // 9. refined: RF[b,:,h*64:+64] = attn[b,h] (64xL) @ VS[b,:,h*64:+64] (Lx64)
    run_gemm<0,0,0>(stream, T1, VS, RF, nullptr, 64, HEADDIM, LTOT,
                    LTOT, DINNER, DINNER, BATCH * NHEADS, NHEADS,
                    (long)NHEADS * 64 * LTOT, 64l * LTOT,
                    (long)LTOT * DINNER, (long)HEADDIM,
                    64l * DINNER, (long)HEADDIM, 1.f);

    // 10. fine: T1[b] = KS[b] (Lx64) @ RF[b] (64x384)
    run_gemm<0,0,0>(stream, KS, RF, T1, nullptr, LTOT, DINNER, DSTATE,
                    DSTATE, DINNER, DINNER, BATCH, 1,
                    (long)LTOT * DSTATE, 0, 64l * DINNER, 0,
                    (long)LTOT * DINNER, 0, 1.f);

    // 11. fused +V*D, LayerNorm, *z (in place on T1)
    ln_mul_kernel<<<ML, 128, 0, stream>>>(T1, XBC, ZXB, Dp, lng, lnb);

    // 12. out = T1 @ W_out
    run_gemm<0,0,0>(stream, T1, W_out, out, nullptr, ML, DMODEL, DINNER,
                    DINNER, DMODEL, DMODEL, 1, 1, 0,0, 0,0, 0,0, 1.f);
}
